// MusicVAEDecoder_55087250539051
// MI455X (gfx1250) — matrix operand order repaired and checked by reading, not compiled, not run on hardware
//
#include <hip/hip_runtime.h>
#include <hip/hip_bf16.h>

typedef __attribute__((ext_vector_type(8)))  __bf16 v8bf;
typedef __attribute__((ext_vector_type(16))) __bf16 v16bf;
typedef __attribute__((ext_vector_type(8)))  float  v8f;

#define UNITS  2048
#define NGATE  8192
#define BATCH  64
#define ZDIM   512
#define TSTEPS 32
#define DIN    90

// Raw split workgroup barrier: hardware sync + compiler ordering, but no
// counter-draining memory fence (we manage ASYNCcnt/DScnt explicitly).
#define WG_BARRIER() asm volatile("s_barrier_signal -1\n\ts_barrier_wait -1" ::: "memory")
// Barrier that first drains this wave's outstanding LDS reads (DScnt), so a
// buffer is never overwritten while another wave's ds_loads are in flight.
#define WG_BARRIER_DS() \
  asm volatile("s_wait_dscnt 0x0\n\ts_barrier_signal -1\n\ts_barrier_wait -1" ::: "memory")

__device__ __forceinline__ float sigmoidf_(float x) { return 1.0f / (1.0f + __expf(-x)); }

// ---------------------------------------------------------------------------
// Transpose fp32 [K][N] (row-major) -> bf16 [N][K] so WMMA B-fragments are
// contiguous 32B per lane.
// ---------------------------------------------------------------------------
__global__ __launch_bounds__(256)
void transpose_f32_to_bf16(const float* __restrict__ src, __hip_bfloat16* __restrict__ dst,
                           int K, int N) {
  __shared__ float tile[32][33];
  const int nb = blockIdx.x * 32, kb = blockIdx.y * 32;
  const int tx = threadIdx.x & 31, ty = threadIdx.x >> 5;
  for (int i = ty; i < 32; i += 8) {
    const int k = kb + i, n = nb + tx;
    tile[i][tx] = (k < K && n < N) ? src[(size_t)k * N + n] : 0.0f;
  }
  __syncthreads();
  for (int i = ty; i < 32; i += 8) {
    const int n = nb + i, k = kb + tx;
    if (n < N && k < K) dst[(size_t)n * K + k] = __float2bfloat16(tile[tx][i]);
  }
}

// ---------------------------------------------------------------------------
// Initial states: st = z @ w_init + b_init, reshape [B][L][2][U];
// which==0 -> h (bf16, buffer 0), which==1 -> c (f32).
// ---------------------------------------------------------------------------
__global__ __launch_bounds__(256)
void init_state_kernel(const float* __restrict__ z, const float* __restrict__ w_init,
                       const float* __restrict__ b_init,
                       __hip_bfloat16* __restrict__ h0, __hip_bfloat16* __restrict__ h1,
                       __hip_bfloat16* __restrict__ h2,
                       float* __restrict__ c0, float* __restrict__ c1, float* __restrict__ c2) {
  const int idx = blockIdx.x * blockDim.x + threadIdx.x;
  if (idx >= BATCH * 12288) return;
  const int b = idx / 12288, col = idx % 12288;
  float acc = b_init[col];
  const float* zr = z + (size_t)b * ZDIM;
  for (int j = 0; j < ZDIM; ++j) acc += zr[j] * w_init[(size_t)j * 12288 + col];
  const int l = col >> 12;        // / (2*U)
  const int rem = col & 4095;
  const int which = rem >> 11;    // / U
  const int u = rem & 2047;
  __hip_bfloat16* hb = (l == 0) ? h0 : (l == 1) ? h1 : h2;
  float* cb = (l == 0) ? c0 : (l == 1) ? c1 : c2;
  if (which == 0) hb[(size_t)b * UNITS + u] = __float2bfloat16(acc);
  else            cb[(size_t)b * UNITS + u] = acc;
}

// ---------------------------------------------------------------------------
// Time-invariant layer-0 input contribution:
// x = [zeros(B,90), z] => xk0 = z @ k0[90:602, :] + b0   (fp32, [64][8192])
// ---------------------------------------------------------------------------
__global__ __launch_bounds__(256)
void xk0_kernel(const float* __restrict__ z, const float* __restrict__ k0,
                const float* __restrict__ b0, float* __restrict__ xk0) {
  const int idx = blockIdx.x * blockDim.x + threadIdx.x;
  if (idx >= BATCH * NGATE) return;
  const int b = idx >> 13, n = idx & (NGATE - 1);
  float acc = b0[n];
  const float* zr = z + (size_t)b * ZDIM;
  for (int j = 0; j < ZDIM; ++j) acc += zr[j] * k0[(size_t)(DIN + j) * NGATE + n];
  xk0[idx] = acc;
}

// ---------------------------------------------------------------------------
// Fused WMMA GEMM + LSTM cell for one (t, layer).
//   zg = h_in @ rk (+ x_in @ k) (+ base | + bias)
//   c' = sig(f)*c + sig(i)*tanh(g) ; h' = sig(o)*tanh(c')
// Grid: 128 blocks (16 units each) x 256 threads (8 waves).
// Wave w: gate = w>>1 (i,f,g,o), M-half = w&1 (rows 0-31 / 32-63).
// 64-wide K-chunks: 64x64 A-tile staged via 2 async global->LDS copies per
// thread (ASYNCcnt), double-buffered; 4 WMMAs per barrier pair; B-fragments
// prefetched one chunk ahead; end-of-iteration barrier drains DScnt first.
// ---------------------------------------------------------------------------
__global__ __launch_bounds__(256)
void lstm_layer_kernel(const __hip_bfloat16* __restrict__ h_in,
                       const __hip_bfloat16* __restrict__ rkT,
                       const __hip_bfloat16* __restrict__ x_in,
                       const __hip_bfloat16* __restrict__ kT,
                       const float* __restrict__ base,   // [64][8192] or null (layer 0: xk0, incl b0)
                       const float* __restrict__ bias,   // [8192] or null (layers 1,2)
                       float* __restrict__ c_state,      // [64][2048] in/out
                       __hip_bfloat16* __restrict__ h_out) {
  __shared__ __bf16 lds_a[2][64][72];    // double-buffered 64x64 A-tile, padded stride
  __shared__ float  lds_zg[4][64][17];   // gate accumulator exchange

  const int tid   = threadIdx.x;
  const int lane  = tid & 31;
  const int wave  = tid >> 5;
  const int gate  = wave >> 1;             // 0..3 = i,f,g,o
  const int mhalf = wave & 1;              // rows 0-31 or 32-63
  const int u0    = blockIdx.x << 4;       // 16 units per block
  const int wn    = gate * UNITS + u0;     // base N-row in [N][K] weights

  // A-fragment addressing (contiguous k = 16*(lane/16) + s, same as B)
  const int arow  = lane & 15;
  const int akoff = (lane >> 4) << 4;      // 0 or 16 elements
  // B-fragment addressing (contiguous k = 16*(lane/16) + s, same as A)
  const int bn    = lane & 15;
  const int bkoff = (lane >> 4) << 4;      // 0 or 16 elements
  // A staging: 256 lanes x 32B covers the 64x64 bf16 tile
  const int srow  = tid >> 2;
  const int sch   = (tid & 3) << 4;        // 16 elements (32B) per thread

  v8f acc[2] = {};                         // 2 M-subtiles of 16x16

  const int npass  = (x_in != nullptr) ? 2 : 1;
  const int nchunk = npass << 5;           // 32 K-chunks (of 64) per pass

  // chunk c: pass = c>>5 (0: h@rk, 1: x@k), K-offset = (c&31)*64
  auto stage_async = [&](int c, int buf) {
    const __bf16* src = (c < 32) ? (const __bf16*)h_in : (const __bf16*)x_in;
    const __bf16* g = src + (size_t)srow * UNITS + ((c & 31) << 6) + sch;
    const unsigned loff = (unsigned)(size_t)&lds_a[buf][srow][sch];
    // INST_OFFSET applies to both global and LDS addresses: reuse regs.
    asm volatile("global_load_async_to_lds_b128 %0, %1, off"
                 :: "v"(loff), "v"(g) : "memory");
    asm volatile("global_load_async_to_lds_b128 %0, %1, off offset:16"
                 :: "v"(loff), "v"(g) : "memory");
  };
  auto load_b = [&](int c, int ks) -> v16bf {
    const __bf16* wt = (c < 32) ? (const __bf16*)rkT : (const __bf16*)kT;
    return *(const v16bf*)(wt + (size_t)(wn + bn) * UNITS + ((c & 31) << 6) + (ks << 5) + bkoff);
  };

  // Pipeline prologue: chunk 0 in flight.
  stage_async(0, 0);
  v16bf bf0 = load_b(0, 0);
  v16bf bf1 = load_b(0, 1);

  for (int c = 0; c < nchunk; ++c) {
    const int cn = (c + 1 < nchunk) ? (c + 1) : c;  // clamped prefetch index
    stage_async(cn, (c + 1) & 1);                   // overwrites buffer last read at c-1
    const v16bf bn0 = load_b(cn, 0);                // B prefetch, one chunk ahead
    const v16bf bn1 = load_b(cn, 1);
    asm volatile("s_wait_asynccnt 0x2" ::: "memory");  // chunk c's 2 copies complete
    WG_BARRIER();                                    // visible to all waves (no loadcnt drain)
    const __bf16* abase = &lds_a[c & 1][0][0];
#pragma unroll
    for (int ks = 0; ks < 2; ++ks) {
#pragma unroll
      for (int s = 0; s < 2; ++s) {
        const int ms = (mhalf << 1) + s;             // this wave's M-subtiles
        const __bf16* ap = abase + (ms * 16 + arow) * 72 + (ks << 5) + akoff;
        const v8bf alo = *(const v8bf*)ap;
        const v8bf ahi = *(const v8bf*)(ap + 8);
        const v16bf afrag = __builtin_shufflevector(
            alo, ahi, 0, 1, 2, 3, 4, 5, 6, 7, 8, 9, 10, 11, 12, 13, 14, 15);
        acc[s] = __builtin_amdgcn_wmma_f32_16x16x32_bf16(
            false, afrag, false, (ks == 0) ? bf0 : bf1, (short)0, acc[s], false, false);
      }
    }
    bf0 = bn0;
    bf1 = bn1;
    WG_BARRIER_DS();  // drain this wave's ds_loads, then sync: no WAR on lds_a
  }

  {  // scatter accumulators: C/D layout -> element r of lane l is M=ms*16+(l>=16?8:0)+r, N=l&15
    const int mbase = (lane >> 4) << 3;
    const int nn = lane & 15;
#pragma unroll
    for (int s = 0; s < 2; ++s) {
      const int ms = (mhalf << 1) + s;
#pragma unroll
      for (int r = 0; r < 8; ++r)
        lds_zg[gate][ms * 16 + mbase + r][nn] = acc[s][r];
    }
  }
  __syncthreads();  // full fence: drains DScnt so the cross-wave exchange is visible

  // Gate epilogue: 1024 (m,u) cells, 4 per thread, uniform control flow.
  for (int e = tid; e < BATCH * 16; e += 256) {
    const int m = e >> 4;
    const int u = e & 15;
    const int gu = u0 + u;
    float zi = lds_zg[0][m][u];
    float zf = lds_zg[1][m][u];
    float zg = lds_zg[2][m][u];
    float zo = lds_zg[3][m][u];
    if (base != nullptr) {
      const float* bp = base + (size_t)m * NGATE;
      zi += bp[gu]; zf += bp[UNITS + gu]; zg += bp[2 * UNITS + gu]; zo += bp[3 * UNITS + gu];
    }
    if (bias != nullptr) {
      zi += bias[gu]; zf += bias[UNITS + gu]; zg += bias[2 * UNITS + gu]; zo += bias[3 * UNITS + gu];
    }
    const float cold = c_state[(size_t)m * UNITS + gu];
    const float cn = sigmoidf_(zf) * cold + sigmoidf_(zi) * tanhf(zg);
    const float hn = sigmoidf_(zo) * tanhf(cn);
    c_state[(size_t)m * UNITS + gu] = cn;
    h_out[(size_t)m * UNITS + gu] = __float2bfloat16(hn);
  }
}

// ---------------------------------------------------------------------------
// logits[t] = h2 @ w_out + b_out, written at [b][t][d] of [B][T][90].
// ---------------------------------------------------------------------------
__global__ __launch_bounds__(128)
void outproj_kernel(const __hip_bfloat16* __restrict__ h2,
                    const __hip_bfloat16* __restrict__ w_outT,  // [90][2048]
                    const float* __restrict__ b_out, float* __restrict__ out, int t) {
  const int idx = blockIdx.x * blockDim.x + threadIdx.x;
  if (idx >= BATCH * DIN) return;
  const int b = idx / DIN, d = idx % DIN;
  const __hip_bfloat16* hr = h2 + (size_t)b * UNITS;
  const __hip_bfloat16* wr = w_outT + (size_t)d * UNITS;
  float acc = 0.0f;
  for (int u = 0; u < UNITS; ++u)
    acc += __bfloat162float(hr[u]) * __bfloat162float(wr[u]);
  out[((size_t)b * TSTEPS + t) * DIN + d] = acc + b_out[d];
}

// ---------------------------------------------------------------------------
extern "C" void kernel_launch(void* const* d_in, const int* in_sizes, int n_in,
                              void* d_out, int out_size, void* d_ws, size_t ws_size,
                              hipStream_t stream) {
  (void)in_sizes; (void)n_in; (void)out_size; (void)ws_size;
  const float* z      = (const float*)d_in[0];
  const float* w_init = (const float*)d_in[1];
  const float* b_init = (const float*)d_in[2];
  const float* k0  = (const float*)d_in[3];
  const float* rk0 = (const float*)d_in[4];
  const float* b0  = (const float*)d_in[5];
  const float* k1  = (const float*)d_in[6];
  const float* rk1 = (const float*)d_in[7];
  const float* b1  = (const float*)d_in[8];
  const float* k2  = (const float*)d_in[9];
  const float* rk2 = (const float*)d_in[10];
  const float* b2  = (const float*)d_in[11];
  const float* w_out = (const float*)d_in[12];
  const float* b_out = (const float*)d_in[13];
  float* out = (float*)d_out;

  // Workspace carve (~173 MB): 5x bf16 weights [8192][2048], w_outT, xk0,
  // double-buffered h (bf16), c (f32).
  char* ws = (char*)d_ws;
  size_t off = 0;
  auto carve = [&](size_t bytes) -> void* {
    void* p = ws + off;
    off += (bytes + 255) & ~(size_t)255;
    return p;
  };
  __hip_bfloat16* rkT[3];
  for (int l = 0; l < 3; ++l) rkT[l] = (__hip_bfloat16*)carve((size_t)NGATE * UNITS * 2);
  __hip_bfloat16* kTw[3];
  kTw[0] = nullptr;
  kTw[1] = (__hip_bfloat16*)carve((size_t)NGATE * UNITS * 2);
  kTw[2] = (__hip_bfloat16*)carve((size_t)NGATE * UNITS * 2);
  __hip_bfloat16* w_outT = (__hip_bfloat16*)carve((size_t)DIN * UNITS * 2);
  float* xk0 = (float*)carve((size_t)BATCH * NGATE * 4);
  __hip_bfloat16* hbuf[3][2];
  for (int l = 0; l < 3; ++l)
    for (int p = 0; p < 2; ++p)
      hbuf[l][p] = (__hip_bfloat16*)carve((size_t)BATCH * UNITS * 2);
  float* cst[3];
  for (int l = 0; l < 3; ++l) cst[l] = (float*)carve((size_t)BATCH * UNITS * 4);

  // Weight conversion fp32 [K][N] -> bf16 [N][K] (L2-resident afterwards).
  dim3 tb(256);
  dim3 tg(NGATE / 32, UNITS / 32);
  transpose_f32_to_bf16<<<tg, tb, 0, stream>>>(rk0, rkT[0], UNITS, NGATE);
  transpose_f32_to_bf16<<<tg, tb, 0, stream>>>(rk1, rkT[1], UNITS, NGATE);
  transpose_f32_to_bf16<<<tg, tb, 0, stream>>>(rk2, rkT[2], UNITS, NGATE);
  transpose_f32_to_bf16<<<tg, tb, 0, stream>>>(k1, kTw[1], UNITS, NGATE);
  transpose_f32_to_bf16<<<tg, tb, 0, stream>>>(k2, kTw[2], UNITS, NGATE);
  transpose_f32_to_bf16<<<dim3((DIN + 31) / 32, UNITS / 32), tb, 0, stream>>>(
      w_out, w_outT, UNITS, DIN);

  init_state_kernel<<<(BATCH * 12288 + 255) / 256, 256, 0, stream>>>(
      z, w_init, b_init, hbuf[0][0], hbuf[1][0], hbuf[2][0], cst[0], cst[1], cst[2]);
  xk0_kernel<<<(BATCH * NGATE + 255) / 256, 256, 0, stream>>>(z, k0, b0, xk0);

  for (int t = 0; t < TSTEPS; ++t) {
    const int rp = t & 1, wp = (t + 1) & 1;
    lstm_layer_kernel<<<UNITS / 16, 256, 0, stream>>>(
        hbuf[0][rp], rkT[0], nullptr, nullptr, xk0, nullptr, cst[0], hbuf[0][wp]);
    lstm_layer_kernel<<<UNITS / 16, 256, 0, stream>>>(
        hbuf[1][rp], rkT[1], hbuf[0][wp], kTw[1], nullptr, b1, cst[1], hbuf[1][wp]);
    lstm_layer_kernel<<<UNITS / 16, 256, 0, stream>>>(
        hbuf[2][rp], rkT[2], hbuf[1][wp], kTw[2], nullptr, b2, cst[2], hbuf[2][wp]);
    outproj_kernel<<<(BATCH * DIN + 127) / 128, 128, 0, stream>>>(
        hbuf[2][wp], w_outT, b_out, out, t);
  }
}
